// GAT_Model_73186242724444
// MI455X (gfx1250) — matrix operand order repaired and checked by reading, not compiled, not run on hardware
//
#include <hip/hip_runtime.h>

// ---------------------------------------------------------------------------
// GAT model for MI455X (gfx1250, wave32, WMMA).
//
// Dense GEMMs (~50 GFLOP) run on v_wmma_f32_16x16x32_bf16.  Per block, the
// 64-column weight slab is staged into LDS once via the CDNA5 async-to-LDS
// path (global_load_async_to_lds_b128 + s_wait_asynccnt), then 8 waves each
// compute a 16x64 strip: A fragments are register double-buffered global
// streams, B fragments are batched ds_load_b128 from the shared slab
// (bank-padded), 4 WMMAs issued back-to-back per k-step.
// Irregular edge-softmax / scatter uses hardware f32 atomics and ordered-uint
// atomicMax.  Everything else is vectorized to b128 accesses (HBM-bound).
// ---------------------------------------------------------------------------

#define NU 100000   // users
#define NF 50000    // foods
#define NE 500000   // edges
#define NP 200000   // label pairs
// HID = 128, H = 4, C = 64, H*C = 512

typedef unsigned short bf16_t;
typedef __attribute__((ext_vector_type(16))) __bf16 v16bf;
typedef __attribute__((ext_vector_type(8)))  float  v8f;

struct U4 { unsigned x, y, z, w; };
union Frag { U4 u[2]; v16bf v; };

__device__ __forceinline__ bf16_t f2bf(float f) {
  unsigned u = __float_as_uint(f);
  u += 0x7FFFu + ((u >> 16) & 1u);          // round-to-nearest-even
  return (bf16_t)(u >> 16);
}
__device__ __forceinline__ float bf2f(bf16_t h) {
  return __uint_as_float((unsigned)h << 16);
}
// unpack one packed u32 (two bf16) -> floats
__device__ __forceinline__ float bfLo(unsigned u) { return __uint_as_float(u << 16); }
__device__ __forceinline__ float bfHi(unsigned u) { return __uint_as_float(u & 0xFFFF0000u); }

// ---------------------------------------------------------------------------
// bf16 GEMM via WMMA:  C[M,N] = A[M,K] * B[K,N] (+bias) (+relu), C in bf16.
// A row-major bf16, Bt = B transposed: Bt[n*K + k] bf16.
// M % 16 == 0 (per-wave 16-row tiles guarded), N % 64 == 0, K in {64,128}.
// ---------------------------------------------------------------------------
template <int K>
__global__ __launch_bounds__(256)
void gemm_bf16_wmma(const bf16_t* __restrict__ A, const bf16_t* __restrict__ Bt,
                    const float* __restrict__ bias, bf16_t* __restrict__ C,
                    int M, int N, int relu)
{
  constexpr int SBS = K + 8;                 // padded LDS stride (bank spread)
  __shared__ bf16_t sB[64 * (128 + 8)];      // max slab: 64 cols x 128 k

  const int tid    = threadIdx.x;
  const int lane   = tid & 31;
  const int wave   = tid >> 5;
  const int laneHi = lane >> 4;              // 0 or 1

  // ---- async-stage 64 columns of Bt into LDS (contiguous 64*K elements) ----
  {
    const bf16_t* src = Bt + (size_t)blockIdx.x * 64 * K;
    constexpr int CPR = K / 8;               // 16B chunks per column
#pragma unroll 4
    for (int j = tid; j < 64 * CPR; j += 256) {
      int colL = j / CPR, kk = j - colL * CPR;
      unsigned ldsOff = (unsigned)(size_t)(sB + colL * SBS + kk * 8);
      unsigned long long g = (unsigned long long)(src + j * 8);
      asm volatile("global_load_async_to_lds_b128 %0, %1, off"
                   :: "v"(ldsOff), "v"(g) : "memory");
    }
    asm volatile("s_wait_asynccnt 0x0" ::: "memory");
  }
  __syncthreads();                            // all waves reach this (no early exit yet)

  const int row0 = blockIdx.y * 128 + wave * 16;
  if (row0 >= M) return;

  const int colL0 = lane & 15;                // local output column (tile 0)
  const bf16_t* Arow = A + (size_t)(row0 + (lane & 15)) * K;

  v8f acc[4] = {};
  // A fragment (16x32 bf16): element e -> k = 16*(e>>3) + 8*laneHi + (e&7)
  Frag aCur, aNxt;
  aCur.u[0] = *(const U4*)(Arow + 8 * laneHi);
  aCur.u[1] = *(const U4*)(Arow + 16 + 8 * laneHi);
  aNxt = aCur;

#pragma unroll
  for (int k0 = 0; k0 < K; k0 += 32) {
    if (k0 + 32 < K) {                        // double-buffer next A slice
      aNxt.u[0] = *(const U4*)(Arow + k0 + 32 + 8 * laneHi);
      aNxt.u[1] = *(const U4*)(Arow + k0 + 48 + 8 * laneHi);
    }
    // batch all 4 B fragments first so the ds_loads clause and the 4 WMMAs
    // issue back-to-back (one dscnt wait instead of four)
    Frag bfr[4];
#pragma unroll
    for (int t = 0; t < 4; ++t) {
      // B fragment (32x16 bf16): element e -> k = 16*(e>>3) + 8*laneHi + (e&7), n = lane&15
      const bf16_t* bsrc = sB + (colL0 + t * 16) * SBS + k0 + 8 * laneHi;
      bfr[t].u[0] = *(const U4*)(bsrc);
      bfr[t].u[1] = *(const U4*)(bsrc + 16);
    }
#pragma unroll
    for (int t = 0; t < 4; ++t) {
      acc[t] = __builtin_amdgcn_wmma_f32_16x16x32_bf16(
          false, aCur.v, false, bfr[t].v, (short)0, acc[t], false, false);
    }
    aCur = aNxt;
  }

  // D layout: VGPR j -> row = row0 + j + 8*laneHi, col = blockIdx.x*64 + colL0 + t*16
#pragma unroll
  for (int t = 0; t < 4; ++t) {
    const int n  = blockIdx.x * 64 + colL0 + t * 16;
    const float bv = bias ? bias[n] : 0.0f;
#pragma unroll
    for (int j = 0; j < 8; ++j) {
      float x = acc[t][j] + bv;
      if (relu) x = fmaxf(x, 0.0f);
      C[(size_t)(row0 + j + 8 * laneHi) * N + n] = f2bf(x);
    }
  }
}

// --------------------------- elementwise helpers ---------------------------

// 8 floats -> 8 bf16 per thread (b128 in, b128 out); n8 = n/8
__global__ void cvt_kernel(const float* __restrict__ src, bf16_t* __restrict__ dst, size_t n8)
{
  size_t i = (size_t)blockIdx.x * blockDim.x + threadIdx.x;
  if (i >= n8) return;
  const float* s = src + i * 8;
  U4 o;
  o.x = ((unsigned)f2bf(s[1]) << 16) | f2bf(s[0]);
  o.y = ((unsigned)f2bf(s[3]) << 16) | f2bf(s[2]);
  o.z = ((unsigned)f2bf(s[5]) << 16) | f2bf(s[4]);
  o.w = ((unsigned)f2bf(s[7]) << 16) | f2bf(s[6]);
  *(U4*)(dst + i * 8) = o;
}

// weight convert + transpose: Wt[n*K + k] = W[k*N + n]
__global__ void cvt_t_kernel(const float* __restrict__ W, bf16_t* __restrict__ Wt, int K, int N)
{
  int i = blockIdx.x * blockDim.x + threadIdx.x;
  if (i >= K * N) return;
  int k = i / N, n = i % N;
  Wt[(size_t)n * K + k] = f2bf(W[i]);
}

// vectorized fill; n4 = n/4 dwords
__global__ void fill_kernel(unsigned* __restrict__ p, unsigned v, size_t n4)
{
  size_t i = (size_t)blockIdx.x * blockDim.x + threadIdx.x;
  if (i < n4) { U4 o = {v, v, v, v}; *(U4*)(p + i * 4) = o; }
}

// s[node*4+h] = sum_c bf2f(hX[node*512 + h*64 + c]) * att[h*64+c]
__global__ void att_score_kernel(const bf16_t* __restrict__ hX, const float* __restrict__ att,
                                 float* __restrict__ s, int n)
{
  int idx = blockIdx.x * blockDim.x + threadIdx.x;
  if (idx >= n * 4) return;
  int node = idx >> 2, h = idx & 3;
  const U4* p8 = (const U4*)(hX + (size_t)node * 512 + h * 64);
  const float* a = att + h * 64;
  float acc = 0.0f;
#pragma unroll
  for (int j = 0; j < 8; ++j) {              // 8 x 16B = 64 bf16
    U4 u = p8[j];
    const float* aj = a + j * 8;
    acc += bfLo(u.x) * aj[0] + bfHi(u.x) * aj[1];
    acc += bfLo(u.y) * aj[2] + bfHi(u.y) * aj[3];
    acc += bfLo(u.z) * aj[4] + bfHi(u.z) * aj[5];
    acc += bfLo(u.w) * aj[6] + bfHi(u.w) * aj[7];
  }
  s[idx] = acc;
}

// ke[h] = sum_c lin_edge[h*64+c] * att_edge[h*64+c]   (edge_attr is [E,1])
__global__ void edge_coef_kernel(const float* __restrict__ lin_e,
                                 const float* __restrict__ att_e, float* __restrict__ ke)
{
  int h = threadIdx.x;
  if (h < 4) {
    float acc = 0.0f;
    for (int c = 0; c < 64; ++c) acc += lin_e[h * 64 + c] * att_e[h * 64 + c];
    ke[h] = acc;
  }
}

__device__ __forceinline__ unsigned ord_enc(float f) {
  unsigned u = __float_as_uint(f);
  return (u & 0x80000000u) ? ~u : (u | 0x80000000u);
}
__device__ __forceinline__ float ord_dec(unsigned u) {
  return __uint_as_float((u & 0x80000000u) ? (u ^ 0x80000000u) : ~u);
}

// alpha = leaky_relu(s_src[src] + s_dst[dst] (+ edge_attr*ke), 0.2); segment max into amax
__global__ void edge_alpha_kernel(const int* __restrict__ sidx, const int* __restrict__ didx,
                                  const float* __restrict__ ssrc, const float* __restrict__ sdst,
                                  const float* __restrict__ eattr, const float* __restrict__ ke,
                                  float* __restrict__ alpha, unsigned* __restrict__ amax, int E)
{
  int idx = blockIdx.x * blockDim.x + threadIdx.x;
  if (idx >= E * 4) return;
  int e = idx >> 2, h = idx & 3;
  int d = didx[e];
  float a = ssrc[(size_t)sidx[e] * 4 + h] + sdst[(size_t)d * 4 + h];
  if (eattr) a += eattr[e] * ke[h];
  a = (a > 0.0f) ? a : 0.2f * a;               // leaky_relu, slope 0.2
  alpha[idx] = a;
  atomicMax(&amax[(size_t)d * 4 + h], ord_enc(a));
}

// ex = exp(alpha - amax[dst]); denom[dst] += ex   (in-place over alpha buffer)
__global__ void edge_exp_kernel(const int* __restrict__ didx, float* __restrict__ alpha,
                                const unsigned* __restrict__ amax, float* __restrict__ denom, int E)
{
  int idx = blockIdx.x * blockDim.x + threadIdx.x;
  if (idx >= E * 4) return;
  int e = idx >> 2, h = idx & 3;
  int d = didx[e];
  float v = __expf(alpha[idx] - ord_dec(amax[(size_t)d * 4 + h]));
  alpha[idx] = v;
  unsafeAtomicAdd(&denom[(size_t)d * 4 + h], v);
}

// agg[dst,h,c] += hs[src,h,c] * ex/(denom+1e-16); one wave per edge, 8 edges/block
__global__ __launch_bounds__(256)
void edge_agg_kernel(const int* __restrict__ sidx, const int* __restrict__ didx,
                     const float* __restrict__ ex, const float* __restrict__ denom,
                     const bf16_t* __restrict__ hs, float* __restrict__ agg, int E)
{
  int e = blockIdx.x * 8 + (threadIdx.x >> 5);
  if (e >= E) return;
  int lane = threadIdx.x & 31;
  int s = sidx[e], d = didx[e];
  const bf16_t* hrow = hs + (size_t)s * 512;
  float* arow = agg + (size_t)d * 256;
#pragma unroll
  for (int i = 0; i < 8; ++i) {
    int v = lane + 32 * i;                    // 0..255 ; h = v>>6
    int h = v >> 6;
    float w = ex[(size_t)e * 4 + h] / (denom[(size_t)d * 4 + h] + 1e-16f);
    unsafeAtomicAdd(&arow[v], bf2f(hrow[v]) * w);
  }
}

// out = relu(bn(mean_h(agg) + conv_bias)) -> bf16
__global__ void finalize_kernel(const float* __restrict__ agg, const float* __restrict__ cbias,
                                const float* __restrict__ g, const float* __restrict__ b,
                                const float* __restrict__ mean, const float* __restrict__ var,
                                bf16_t* __restrict__ out, int n)
{
  int idx = blockIdx.x * blockDim.x + threadIdx.x;
  if (idx >= n * 64) return;
  int node = idx >> 6, c = idx & 63;
  const float* p = agg + (size_t)node * 256;
  float x = 0.25f * (p[c] + p[64 + c] + p[128 + c] + p[192 + c]) + cbias[c];
  x = (x - mean[c]) * rsqrtf(var[c] + 1e-5f) * g[c] + b[c];
  out[(size_t)node * 64 + c] = f2bf(fmaxf(x, 0.0f));
}

// comb[p,0:64] = user_out[label_user[p]], comb[p,64:128] = food_out[label_food[p]]
// one 16B chunk per thread: P * 16 chunks
__global__ void gather_kernel(const int* __restrict__ lu, const int* __restrict__ lf,
                              const bf16_t* __restrict__ uo, const bf16_t* __restrict__ fo,
                              bf16_t* __restrict__ comb, int P)
{
  int idx = blockIdx.x * blockDim.x + threadIdx.x;
  if (idx >= P * 16) return;
  int p = idx >> 4, ch = idx & 15;
  const bf16_t* src = (ch < 8) ? (uo + (size_t)lu[p] * 64 + ch * 8)
                               : (fo + (size_t)lf[p] * 64 + (ch - 8) * 8);
  *(U4*)(comb + (size_t)idx * 8) = *(const U4*)src;
}

// out[p] = sigmoid(h1[p,:] . W2 + b2)
__global__ void final_dot_kernel(const bf16_t* __restrict__ h1, const float* __restrict__ W2,
                                 const float* __restrict__ b2, float* __restrict__ out, int P)
{
  int p = blockIdx.x * blockDim.x + threadIdx.x;
  if (p >= P) return;
  const U4* r8 = (const U4*)(h1 + (size_t)p * 128);
  float s = b2[0];
#pragma unroll
  for (int j = 0; j < 16; ++j) {             // 16 x 16B = 128 bf16
    U4 u = r8[j];
    const float* wj = W2 + j * 8;
    s += bfLo(u.x) * wj[0] + bfHi(u.x) * wj[1];
    s += bfLo(u.y) * wj[2] + bfHi(u.y) * wj[3];
    s += bfLo(u.z) * wj[4] + bfHi(u.z) * wj[5];
    s += bfLo(u.w) * wj[6] + bfHi(u.w) * wj[7];
  }
  out[p] = 1.0f / (1.0f + __expf(-s));
}

// ---------------------------------------------------------------------------

static inline unsigned cdiv(size_t a, unsigned b) { return (unsigned)((a + b - 1) / b); }

extern "C" void kernel_launch(void* const* d_in, const int* in_sizes, int n_in,
                              void* d_out, int out_size, void* d_ws, size_t ws_size,
                              hipStream_t stream)
{
  (void)in_sizes; (void)n_in; (void)out_size; (void)ws_size;

  const float* x_user = (const float*)d_in[0];
  const float* x_food = (const float*)d_in[1];
  const float* eattr  = (const float*)d_in[2];
  const float* puW = (const float*)d_in[3];
  const float* pub = (const float*)d_in[4];
  const float* pfW = (const float*)d_in[5];
  const float* pfb = (const float*)d_in[6];
  const float* elsW = (const float*)d_in[7];
  const float* eldW = (const float*)d_in[8];
  const float* eas  = (const float*)d_in[9];
  const float* ead  = (const float*)d_in[10];
  const float* ele  = (const float*)d_in[11];
  const float* eae  = (const float*)d_in[12];
  const float* ebias = (const float*)d_in[13];
  const float* rlsW = (const float*)d_in[14];
  const float* rldW = (const float*)d_in[15];
  const float* ras  = (const float*)d_in[16];
  const float* rad  = (const float*)d_in[17];
  const float* rbias = (const float*)d_in[18];
  const float* bug = (const float*)d_in[19];
  const float* bub = (const float*)d_in[20];
  const float* bum = (const float*)d_in[21];
  const float* buv = (const float*)d_in[22];
  const float* bfg = (const float*)d_in[23];
  const float* bfbeta = (const float*)d_in[24];
  const float* bfm = (const float*)d_in[25];
  const float* bfv = (const float*)d_in[26];
  const float* dW1 = (const float*)d_in[27];
  const float* db1 = (const float*)d_in[28];
  const float* dW2 = (const float*)d_in[29];
  const float* db2 = (const float*)d_in[30];
  const int* esrc = (const int*)d_in[31];
  const int* edst = (const int*)d_in[32];
  const int* lu   = (const int*)d_in[33];
  const int* lf   = (const int*)d_in[34];
  float* out = (float*)d_out;

  // ---- workspace layout (256B-aligned slabs, big buffers reused) ----
  char* ws = (char*)d_ws;
  size_t off = 0;
  auto take = [&](size_t bytes) -> char* {
    char* p = ws + off;
    off = (off + bytes + 255) & ~(size_t)255;
    return p;
  };
  bf16_t* xu_b  = (bf16_t*)take((size_t)NU * 64 * 2);
  bf16_t* xf_b  = (bf16_t*)take((size_t)NF * 128 * 2);
  bf16_t* WtPU  = (bf16_t*)take((size_t)64 * 128 * 2);
  bf16_t* WtPF  = (bf16_t*)take((size_t)128 * 128 * 2);
  bf16_t* WtELS = (bf16_t*)take((size_t)128 * 512 * 2);
  bf16_t* WtELD = (bf16_t*)take((size_t)128 * 512 * 2);
  bf16_t* WtRLS = (bf16_t*)take((size_t)128 * 512 * 2);
  bf16_t* WtRLD = (bf16_t*)take((size_t)128 * 512 * 2);
  bf16_t* WtD1  = (bf16_t*)take((size_t)128 * 128 * 2);
  bf16_t* hu_b  = (bf16_t*)take((size_t)NU * 128 * 2);
  bf16_t* hf_b  = (bf16_t*)take((size_t)NF * 128 * 2);
  bf16_t* hs_b  = (bf16_t*)take((size_t)NU * 512 * 2);   // per-conv src proj; later comb
  bf16_t* hd_b  = (bf16_t*)take((size_t)NU * 512 * 2);   // per-conv dst proj; later h1
  float*  s_src = (float*)take((size_t)NU * 4 * 4);
  float*  s_dst = (float*)take((size_t)NU * 4 * 4);
  float*  ke    = (float*)take(256);
  float*  alpha = (float*)take((size_t)NE * 4 * 4);      // alpha, then ex (in place)
  unsigned* amax = (unsigned*)take((size_t)NU * 4 * 4);
  float*  denom = (float*)take((size_t)NU * 4 * 4);
  float*  agg   = (float*)take((size_t)NU * 256 * 4);
  bf16_t* uo_b  = (bf16_t*)take((size_t)NU * 64 * 2);
  bf16_t* fo_b  = (bf16_t*)take((size_t)NF * 64 * 2);
  bf16_t* comb  = hs_b;                                  // reuse (NP*128 <= NU*512)
  bf16_t* h1    = hd_b;                                  // reuse

  // ---- bf16 conversions ----
  cvt_kernel<<<cdiv((size_t)NU * 64 / 8, 256), 256, 0, stream>>>(x_user, xu_b, (size_t)NU * 64 / 8);
  cvt_kernel<<<cdiv((size_t)NF * 128 / 8, 256), 256, 0, stream>>>(x_food, xf_b, (size_t)NF * 128 / 8);
  cvt_t_kernel<<<cdiv(64 * 128, 256), 256, 0, stream>>>(puW, WtPU, 64, 128);
  cvt_t_kernel<<<cdiv(128 * 128, 256), 256, 0, stream>>>(pfW, WtPF, 128, 128);
  cvt_t_kernel<<<cdiv(128 * 512, 256), 256, 0, stream>>>(elsW, WtELS, 128, 512);
  cvt_t_kernel<<<cdiv(128 * 512, 256), 256, 0, stream>>>(eldW, WtELD, 128, 512);
  cvt_t_kernel<<<cdiv(128 * 512, 256), 256, 0, stream>>>(rlsW, WtRLS, 128, 512);
  cvt_t_kernel<<<cdiv(128 * 512, 256), 256, 0, stream>>>(rldW, WtRLD, 128, 512);
  cvt_t_kernel<<<cdiv(128 * 128, 256), 256, 0, stream>>>(dW1, WtD1, 128, 128);

  // ---- node projections (WMMA) ----
  gemm_bf16_wmma<64><<<dim3(2, cdiv(NU, 128)), 256, 0, stream>>>(xu_b, WtPU, pub, hu_b, NU, 128, 0);
  gemm_bf16_wmma<128><<<dim3(2, cdiv(NF, 128)), 256, 0, stream>>>(xf_b, WtPF, pfb, hf_b, NF, 128, 0);

  // ================= eats conv: user -> food =================
  gemm_bf16_wmma<128><<<dim3(8, cdiv(NU, 128)), 256, 0, stream>>>(hu_b, WtELS, nullptr, hs_b, NU, 512, 0);
  gemm_bf16_wmma<128><<<dim3(8, cdiv(NF, 128)), 256, 0, stream>>>(hf_b, WtELD, nullptr, hd_b, NF, 512, 0);
  att_score_kernel<<<cdiv((size_t)NU * 4, 256), 256, 0, stream>>>(hs_b, eas, s_src, NU);
  att_score_kernel<<<cdiv((size_t)NF * 4, 256), 256, 0, stream>>>(hd_b, ead, s_dst, NF);
  edge_coef_kernel<<<1, 32, 0, stream>>>(ele, eae, ke);
  fill_kernel<<<cdiv((size_t)NF, 256), 256, 0, stream>>>(amax, 0u, (size_t)NF);
  fill_kernel<<<cdiv((size_t)NF, 256), 256, 0, stream>>>((unsigned*)denom, 0u, (size_t)NF);
  fill_kernel<<<cdiv((size_t)NF * 64, 256), 256, 0, stream>>>((unsigned*)agg, 0u, (size_t)NF * 64);
  edge_alpha_kernel<<<cdiv((size_t)NE * 4, 256), 256, 0, stream>>>(esrc, edst, s_src, s_dst, eattr, ke, alpha, amax, NE);
  edge_exp_kernel<<<cdiv((size_t)NE * 4, 256), 256, 0, stream>>>(edst, alpha, amax, denom, NE);
  edge_agg_kernel<<<cdiv(NE, 8), 256, 0, stream>>>(esrc, edst, alpha, denom, hs_b, agg, NE);
  finalize_kernel<<<cdiv((size_t)NF * 64, 256), 256, 0, stream>>>(agg, ebias, bfg, bfbeta, bfm, bfv, fo_b, NF);

  // ================= rev conv: food -> user (reversed edges) =================
  gemm_bf16_wmma<128><<<dim3(8, cdiv(NF, 128)), 256, 0, stream>>>(hf_b, WtRLS, nullptr, hs_b, NF, 512, 0);
  gemm_bf16_wmma<128><<<dim3(8, cdiv(NU, 128)), 256, 0, stream>>>(hu_b, WtRLD, nullptr, hd_b, NU, 512, 0);
  att_score_kernel<<<cdiv((size_t)NF * 4, 256), 256, 0, stream>>>(hs_b, ras, s_src, NF);
  att_score_kernel<<<cdiv((size_t)NU * 4, 256), 256, 0, stream>>>(hd_b, rad, s_dst, NU);
  fill_kernel<<<cdiv((size_t)NU, 256), 256, 0, stream>>>(amax, 0u, (size_t)NU);
  fill_kernel<<<cdiv((size_t)NU, 256), 256, 0, stream>>>((unsigned*)denom, 0u, (size_t)NU);
  fill_kernel<<<cdiv((size_t)NU * 64, 256), 256, 0, stream>>>((unsigned*)agg, 0u, (size_t)NU * 64);
  edge_alpha_kernel<<<cdiv((size_t)NE * 4, 256), 256, 0, stream>>>(edst, esrc, s_src, s_dst, nullptr, nullptr, alpha, amax, NE);
  edge_exp_kernel<<<cdiv((size_t)NE * 4, 256), 256, 0, stream>>>(esrc, alpha, amax, denom, NE);
  edge_agg_kernel<<<cdiv(NE, 8), 256, 0, stream>>>(edst, esrc, alpha, denom, hs_b, agg, NE);
  finalize_kernel<<<cdiv((size_t)NU * 64, 256), 256, 0, stream>>>(agg, rbias, bug, bub, bum, buv, uo_b, NU);

  // ================= decoder =================
  gather_kernel<<<cdiv((size_t)NP * 16, 256), 256, 0, stream>>>(lu, lf, uo_b, fo_b, comb, NP);
  gemm_bf16_wmma<128><<<dim3(2, cdiv(NP, 128)), 256, 0, stream>>>(comb, WtD1, db1, h1, NP, 128, 1);
  final_dot_kernel<<<cdiv((size_t)NP, 256), 256, 0, stream>>>(h1, dW2, db2, out, NP);
}
